// SAModule_63299228009289
// MI455X (gfx1250) — matrix operand order repaired and checked by reading, not compiled, not run on hardware
//
#include <hip/hip_runtime.h>
#include <math.h>

#define B_    4
#define N_    8192
#define CIN   64
#define COUT  128
#define S_    2048
#define KNBR  64
#define DIN   67          // C_IN + 3
#define R2    0.16f       // 0.4^2
#define CAP   1024        // candidate buffer per query

// Swizzled weight-fragment element counts (see wconv_kernel)
#define W1F_N 6144        // 12 tiles (3 ks x 4 nt) x 512
#define W2F_N 4096        // 8 tiles  (2 ks x 4 nt) x 512
#define W3F_N 8192        // 16 tiles (2 ks x 8 nt) x 512
#define WF_N  (W1F_N + W2F_N + W3F_N)   // 18432 elements = 36864 bytes

typedef __attribute__((ext_vector_type(16))) __bf16          v16bf;
typedef __attribute__((ext_vector_type(16))) unsigned short  v16us;
typedef __attribute__((ext_vector_type(8)))  float           v8f;
typedef __attribute__((ext_vector_type(4)))  int             v4i;

__device__ __forceinline__ unsigned short f2bf(float f) {
    unsigned u = __float_as_uint(f);
    u += 0x7FFFu + ((u >> 16) & 1u);        // round-to-nearest-even
    return (unsigned short)(u >> 16);
}

// A-fragment intra-lane K offset for element i, lane-half h (16-bit A, 16x32)
__device__ __forceinline__ int a_kl(int i, int h) {
    int v = i >> 1;
    return (v < 4 ? 2 * v : 16 + 2 * (v - 4)) + h * 8 + (i & 1);
}

__device__ __forceinline__ v16bf fragl(const unsigned short* p) {
    return __builtin_bit_cast(v16bf, *(const v16us*)p);
}

// ------------------------------------------------------- weights -> bf16 fragments
// B-fragment layout (32x16 bf16): lane = col n (lo 4 bits), lane>=16 -> K+8,
// element i -> k = ks*32 + (lane>>4)*8 + i + (i>=8 ? 8 : 0).  Tile index: ks*NT + nt.
__global__ void wconv_kernel(const float* __restrict__ W1,
                             const float* __restrict__ W2,
                             const float* __restrict__ W3,
                             unsigned short* __restrict__ wpAll) {
    int i = blockIdx.x * blockDim.x + threadIdx.x;
    if (i >= WF_N) return;
    float val;
    if (i < W1F_N) {
        int o = i, T = o >> 9, ks = T >> 2, nt = T & 3;
        int l = (o >> 4) & 31, e = o & 15;
        int n = nt * 16 + (l & 15);
        int k = ks * 32 + (l >> 4) * 8 + e + (e >= 8 ? 8 : 0);
        val = (k < DIN) ? W1[k * 64 + n] : 0.0f;
    } else if (i < W1F_N + W2F_N) {
        int o = i - W1F_N, T = o >> 9, ks = T >> 2, nt = T & 3;
        int l = (o >> 4) & 31, e = o & 15;
        int n = nt * 16 + (l & 15);
        int k = ks * 32 + (l >> 4) * 8 + e + (e >= 8 ? 8 : 0);
        val = W2[k * 64 + n];
    } else {
        int o = i - W1F_N - W2F_N, T = o >> 9, ks = T >> 3, nt = T & 7;
        int l = (o >> 4) & 31, e = o & 15;
        int n = nt * 16 + (l & 15);
        int k = ks * 32 + (l >> 4) * 8 + e + (e >= 8 ? 8 : 0);
        val = W3[k * 128 + n];
    }
    wpAll[i] = f2bf(val);
}

// ---------------------------------------------------------------- FPS
__global__ __launch_bounds__(1024) void fps_kernel(const float* __restrict__ pos,
                                                   float* __restrict__ pos_s) {
    const int b    = blockIdx.x;
    const int tid  = threadIdx.x;
    const int lane = tid & 31;
    const int wv   = tid >> 5;

    __shared__ float rv[32];
    __shared__ int   ri[32];
    __shared__ int   s_last;

    float px[8], py[8], pz[8], md[8];
    int   gi[8];
#pragma unroll
    for (int i = 0; i < 8; i++) {
        int j = tid + i * 1024;
        gi[i] = j;
        px[i] = pos[(b * N_ + j) * 3 + 0];
        py[i] = pos[(b * N_ + j) * 3 + 1];
        pz[i] = pos[(b * N_ + j) * 3 + 2];
        md[i] = 3.4e38f;
    }
    if (tid == 0) {
        s_last = 0;
        pos_s[(b * S_) * 3 + 0] = pos[(b * N_) * 3 + 0];
        pos_s[(b * S_) * 3 + 1] = pos[(b * N_) * 3 + 1];
        pos_s[(b * S_) * 3 + 2] = pos[(b * N_) * 3 + 2];
    }
    __syncthreads();
    int last = s_last;

    for (int t = 1; t < S_; t++) {
        float lx = pos[(b * N_ + last) * 3 + 0];
        float ly = pos[(b * N_ + last) * 3 + 1];
        float lz = pos[(b * N_ + last) * 3 + 2];
        float bv = -1.0f; int bi = 0;
#pragma unroll
        for (int i = 0; i < 8; i++) {
            float dx = px[i] - lx, dy = py[i] - ly, dz = pz[i] - lz;
            float d  = dx * dx + dy * dy + dz * dz;
            md[i] = fminf(md[i], d);
            if (md[i] > bv || (md[i] == bv && gi[i] < bi)) { bv = md[i]; bi = gi[i]; }
        }
#pragma unroll
        for (int off = 16; off >= 1; off >>= 1) {
            float ov = __shfl_xor(bv, off, 32);
            int   oi = __shfl_xor(bi, off, 32);
            if (ov > bv || (ov == bv && oi < bi)) { bv = ov; bi = oi; }
        }
        if (lane == 0) { rv[wv] = bv; ri[wv] = bi; }
        __syncthreads();
        if (wv == 0) {
            float v  = rv[lane];
            int   ix = ri[lane];
#pragma unroll
            for (int off = 16; off >= 1; off >>= 1) {
                float ov = __shfl_xor(v, off, 32);
                int   oi = __shfl_xor(ix, off, 32);
                if (ov > v || (ov == v && oi < ix)) { v = ov; ix = oi; }
            }
            if (lane == 0) {
                s_last = ix;
                pos_s[(b * S_ + t) * 3 + 0] = pos[(b * N_ + ix) * 3 + 0];
                pos_s[(b * S_ + t) * 3 + 1] = pos[(b * N_ + ix) * 3 + 1];
                pos_s[(b * S_ + t) * 3 + 2] = pos[(b * N_ + ix) * 3 + 2];
            }
        }
        __syncthreads();
        last = s_last;
    }
}

// ---------------------------------------------------------------- radius kNN
__global__ __launch_bounds__(256) void nbr_kernel(const float* __restrict__ pos,
                                                  const float* __restrict__ pos_s,
                                                  int* __restrict__ nbr,
                                                  int* __restrict__ nvalid) {
    __shared__ float cd[CAP];
    __shared__ int   ci[CAP];
    __shared__ int   scnt;
    __shared__ float rv[8];
    __shared__ int   ri[8];
    const int tid = threadIdx.x, lane = tid & 31, wv = tid >> 5;

    for (int qi = 0; qi < 8; qi++) {
        int q = blockIdx.x * 8 + qi;
        int b = q / S_;
        float qx = pos_s[q * 3 + 0], qy = pos_s[q * 3 + 1], qz = pos_s[q * 3 + 2];
        if (tid == 0) scnt = 0;
        __syncthreads();
        for (int j = tid; j < N_; j += 256) {
            float dx = pos[(b * N_ + j) * 3 + 0] - qx;
            float dy = pos[(b * N_ + j) * 3 + 1] - qy;
            float dz = pos[(b * N_ + j) * 3 + 2] - qz;
            float d  = dx * dx + dy * dy + dz * dz;
            if (d <= R2) {
                int s = atomicAdd(&scnt, 1);
                if (s < CAP) { cd[s] = d; ci[s] = j; }
            }
        }
        __syncthreads();
        int n = scnt; if (n > CAP) n = CAP;
        if (n <= KNBR) {
            if (tid < n)   nbr[q * KNBR + tid] = ci[tid];
            if (tid == 0)  nvalid[q] = n;
        } else {
            for (int r = 0; r < KNBR; r++) {
                float bv = 3.4e38f; int bi = -1;
                for (int s = tid; s < n; s += 256) {
                    float v = cd[s];
                    if (v < bv) { bv = v; bi = s; }
                }
#pragma unroll
                for (int off = 16; off >= 1; off >>= 1) {
                    float ov = __shfl_xor(bv, off, 32);
                    int   oi = __shfl_xor(bi, off, 32);
                    if (ov < bv || (ov == bv && oi < bi)) { bv = ov; bi = oi; }
                }
                if (lane == 0) { rv[wv] = bv; ri[wv] = bi; }
                __syncthreads();
                if (tid == 0) {
                    float v = rv[0]; int ix = ri[0];
                    for (int w = 1; w < 8; w++)
                        if (rv[w] < v || (rv[w] == v && ri[w] < ix)) { v = rv[w]; ix = ri[w]; }
                    cd[ix] = 3.4e38f;
                    nbr[q * KNBR + r] = ci[ix];
                }
                __syncthreads();
            }
            if (tid == 0) nvalid[q] = KNBR;
        }
        __syncthreads();
    }
}

// ---------------------------------------------------------------- gather + MLP + max
__global__ __launch_bounds__(128) void mlp_kernel(const float* __restrict__ x,
                                                  const float* __restrict__ pos,
                                                  const float* __restrict__ pos_s,
                                                  const int* __restrict__ nbr,
                                                  const int* __restrict__ nvalid,
                                                  const unsigned short* __restrict__ wpAll,
                                                  const float* __restrict__ b1,
                                                  const float* __restrict__ b2,
                                                  const float* __restrict__ b3,
                                                  float* __restrict__ out) {
    // All buffers hold data in WMMA-fragment order: tile*512 + lane*16 + elem.
    __shared__ __attribute__((aligned(32))) unsigned short sWf [WF_N];   // 36864 B
    __shared__ __attribute__((aligned(32))) unsigned short sAf [6144];   // 12 tiles (4mt x 3ks)
    __shared__ __attribute__((aligned(32))) unsigned short sH1f[4096];   // 8 tiles  (4mt x 2ks)
    __shared__ int snbr[KNBR];
    __shared__ int snv;

    unsigned short* sW1f = sWf;                 // 12 tiles
    unsigned short* sW2f = sWf + W1F_N;         // 8 tiles
    unsigned short* sW3f = sWf + W1F_N + W2F_N; // 16 tiles
    unsigned short* sH2f = sAf;                 // alias: sA dead after GEMM1

    const int tid = threadIdx.x, lane = tid & 31, wv = tid >> 5;

    // ---- weight fragments: global -> LDS (async path if toolchain exposes it)
#if defined(__has_builtin) && __has_builtin(__builtin_amdgcn_global_load_async_to_lds_b128)
    {
        typedef __attribute__((address_space(1))) v4i* g4p;
        typedef __attribute__((address_space(3))) v4i* l4p;
        g4p g = (g4p)wpAll;
        l4p l = (l4p)sWf;
        for (int c = tid; c < WF_N / 16; c += 128)
            __builtin_amdgcn_global_load_async_to_lds_b128(g + c, l + c, 0, 0);
#if __has_builtin(__builtin_amdgcn_s_wait_asynccnt)
        __builtin_amdgcn_s_wait_asynccnt(0);
#else
        asm volatile("s_wait_asynccnt 0" ::: "memory");
#endif
    }
#else
    for (int c = tid; c < WF_N / 16; c += 128)
        *(v16us*)&sWf[c * 16] = *(const v16us*)&wpAll[c * 16];
#endif

    for (int qi = 0; qi < 4; qi++) {
        int q = blockIdx.x * 4 + qi;
        int b = q / S_;
        if (tid == 0) snv = nvalid[q];
        __syncthreads();
        int nv = snv;
        if (tid < KNBR) snbr[tid] = (tid < nv) ? nbr[q * KNBR + tid] : 0;
        __syncthreads();

        float qx = pos_s[q * 3 + 0], qy = pos_s[q * 3 + 1], qz = pos_s[q * 3 + 2];

        // Build A directly in fragment order (K padded 67->96 with zeros).
        for (int e = tid; e < 6144; e += 128) {
            int T = e >> 9;                 // tile = mt*3 + ks
            int l = (e >> 4) & 31, i = e & 15;
            int mt = T / 3, ks = T - mt * 3;
            int m  = mt * 16 + (l & 15);
            int k  = ks * 32 + a_kl(i, l >> 4);
            int id = snbr[m];
            float v;
            if (k < CIN)       v = x[(b * N_ + id) * CIN + k];
            else if (k == 64)  v = pos[(b * N_ + id) * 3 + 0] - qx;
            else if (k == 65)  v = pos[(b * N_ + id) * 3 + 1] - qy;
            else if (k == 66)  v = pos[(b * N_ + id) * 3 + 2] - qz;
            else               v = 0.0f;
            sAf[e] = f2bf(v);
        }
        __syncthreads();

        const int half = lane >> 4;

        // GEMM1: (64x96)@(96x64) + b1, relu.  wave wv -> N-tile wv.
        {
            int   n    = wv * 16 + (lane & 15);
            float bias = b1[n];
            // target address pieces for H1 fragment store (GEMM2 A-layout)
            int kl  = n & 31, h2 = (kl >> 3) & 1;
            int klp = kl - 8 * h2;
            int i2  = (klp < 8) ? klp : klp - 8;
            int ks2 = n >> 5;
            for (int mt = 0; mt < 4; mt++) {
                v8f c = {};
#pragma unroll
                for (int ks = 0; ks < 3; ks++) {
                    v16bf a  = fragl(&sAf [(mt * 3 + ks) * 512 + lane * 16]);
                    v16bf bb = fragl(&sW1f[(ks * 4 + wv) * 512 + lane * 16]);
                    c = __builtin_amdgcn_wmma_f32_16x16x32_bf16(false, a, false, bb,
                                                                (short)0, c, false, false);
                }
                int base = (mt * 2 + ks2) * 512 + (h2 * 16) * 16 + i2;
#pragma unroll
                for (int r = 0; r < 8; r++) {
                    float v = c[r] + bias;
                    v = v > 0.0f ? v : 0.0f;
                    sH1f[base + (half * 8 + r) * 16] = f2bf(v);
                }
            }
        }
        __syncthreads();

        // GEMM2: (64x64)@(64x64) + b2, relu.
        {
            int   n    = wv * 16 + (lane & 15);
            float bias = b2[n];
            int kl  = n & 31, h2 = (kl >> 3) & 1;
            int klp = kl - 8 * h2;
            int i2  = (klp < 8) ? klp : klp - 8;
            int ks2 = n >> 5;
            for (int mt = 0; mt < 4; mt++) {
                v8f c = {};
#pragma unroll
                for (int ks = 0; ks < 2; ks++) {
                    v16bf a  = fragl(&sH1f[(mt * 2 + ks) * 512 + lane * 16]);
                    v16bf bb = fragl(&sW2f[(ks * 4 + wv) * 512 + lane * 16]);
                    c = __builtin_amdgcn_wmma_f32_16x16x32_bf16(false, a, false, bb,
                                                                (short)0, c, false, false);
                }
                int base = (mt * 2 + ks2) * 512 + (h2 * 16) * 16 + i2;
#pragma unroll
                for (int r = 0; r < 8; r++) {
                    float v = c[r] + bias;
                    v = v > 0.0f ? v : 0.0f;
                    sH2f[base + (half * 8 + r) * 16] = f2bf(v);
                }
            }
        }
        __syncthreads();

        // GEMM3: (64x64)@(64x128) + b3, masked max over K rows -> out[q][128]
        for (int p = 0; p < 2; p++) {
            int nt = wv * 2 + p;
            int n  = nt * 16 + (lane & 15);
            float cmax = -3.4e38f;
            for (int mt = 0; mt < 4; mt++) {
                v8f c = {};
#pragma unroll
                for (int ks = 0; ks < 2; ks++) {
                    v16bf a  = fragl(&sH2f[(mt * 2 + ks) * 512 + lane * 16]);
                    v16bf bb = fragl(&sW3f[(ks * 8 + nt) * 512 + lane * 16]);
                    c = __builtin_amdgcn_wmma_f32_16x16x32_bf16(false, a, false, bb,
                                                                (short)0, c, false, false);
                }
#pragma unroll
                for (int r = 0; r < 8; r++) {
                    int k = mt * 16 + half * 8 + r;
                    if (k < nv) cmax = fmaxf(cmax, c[r]);
                }
            }
            cmax = fmaxf(cmax, __shfl_xor(cmax, 16, 32));
            if (lane < 16)
                out[q * COUT + n] = (nv > 0) ? (cmax + b3[n]) : 0.0f;
        }
        __syncthreads();
    }
}

// ---------------------------------------------------------------- host entry
extern "C" void kernel_launch(void* const* d_in, const int* in_sizes, int n_in,
                              void* d_out, int out_size, void* d_ws, size_t ws_size,
                              hipStream_t stream) {
    const float* x   = (const float*)d_in[0];
    const float* pos = (const float*)d_in[1];
    const float* W1  = (const float*)d_in[2];
    const float* b1  = (const float*)d_in[3];
    const float* W2  = (const float*)d_in[4];
    const float* b2  = (const float*)d_in[5];
    const float* W3  = (const float*)d_in[6];
    const float* b3  = (const float*)d_in[7];

    float* out   = (float*)d_out;
    float* pos_s = out + (size_t)B_ * S_ * COUT;   // second output, concatenated

    char* ws = (char*)d_ws;
    int* nvalid = (int*)ws;                                   // 32 KB
    int* nbr    = (int*)(ws + 32768);                         // 2 MB
    unsigned short* wpAll = (unsigned short*)(ws + 32768 + 2097152); // 32B-aligned, 36864 B

    wconv_kernel<<<(WF_N + 255) / 256, 256, 0, stream>>>(W1, W2, W3, wpAll);
    fps_kernel  <<<B_, 1024, 0, stream>>>(pos, pos_s);
    nbr_kernel  <<<(B_ * S_) / 8, 256, 0, stream>>>(pos, pos_s, nbr, nvalid);
    mlp_kernel  <<<(B_ * S_) / 4, 128, 0, stream>>>(x, pos, pos_s, nbr, nvalid,
                                                    wpAll, b1, b2, b3, out);
}
